// GroupedQueryAttention_56873956934227
// MI455X (gfx1250) — matrix operand order repaired and checked by reading, not compiled, not run on hardware
//
#include <hip/hip_runtime.h>
#include <math.h>

// ---------------- problem constants ----------------
static constexpr int B_  = 2;
static constexpr int S_  = 2048;
static constexpr int D_  = 2048;
static constexpr int H_  = 32;
static constexpr int G_  = 8;
static constexpr int HD_ = 64;

typedef __attribute__((ext_vector_type(16))) __bf16   v16bf;
typedef __attribute__((ext_vector_type(8)))  float    v8f;
typedef __attribute__((ext_vector_type(4)))  unsigned u32x4;
typedef __attribute__((ext_vector_type(8)))  int      i32x8;
typedef __attribute__((ext_vector_type(4)))  int      i32x4;

// ---------------- CDNA5 async / TDM helpers ----------------
__device__ inline void async_load_b128(unsigned lds_off, const void* gaddr) {
  // DMA 16B from global to LDS, tracked by ASYNCcnt (no VGPR round-trip).
  asm volatile("global_load_async_to_lds_b128 %0, %1, off"
               :: "v"(lds_off), "v"(gaddr) : "memory");
}
__device__ inline void wait_async0() {
  asm volatile("s_wait_asynccnt 0x0" ::: "memory");
}

// TDM: 2D tile load, 32 rows x 128B, bf16 viewed as 4B units (dim0 = 32 units).
// LDS pad: +16B per row (pad_interval = 32 dwords -> code 4, pad_amount = 4
// dwords -> code 3) reproducing the [HD_+8] bf16 row stride of 144B.
__device__ inline void tdm_load_tile32(unsigned lds_addr, const __bf16* gptr) {
  unsigned long long ga = (unsigned long long)(uintptr_t)gptr;
  u32x4 g0;
  g0[0] = 1u;                                             // count=1, user mode
  g0[1] = lds_addr;                                       // LDS byte address
  g0[2] = (unsigned)(ga & 0xffffffffu);                   // global addr lo
  g0[3] = (unsigned)((ga >> 32) & 0x01ffffffu) | (2u << 30); // addr[56:32]|type=2
  i32x8 g1;
  g1[0] = (2 << 16)        // data_size = 4B
        | (1 << 20)        // pad_enable
        | (4 << 22)        // pad_interval: 32 DWORDs (one 128B row)
        | (3 << 25);       // pad_amount: 4 DWORDs (16B)
  g1[1] = (32 << 16);      // tensor_dim0[15:0] = 32 units (128B row)
  g1[2] = (S_ & 0xffff) << 16; // tensor_dim1[15:0] = S rows
  g1[3] = (32 << 16);      // tensor_dim1 hi = 0 | tile_dim0 = 32 units
  g1[4] = 32;              // tile_dim1 = 32 rows, tile_dim2 = 0
  g1[5] = 32;              // tensor_dim0_stride = 32 units
  g1[6] = (32 << 16);      // tensor_dim1_stride[15:0] = 32 units
  g1[7] = 0;
  i32x4 z4 = {0, 0, 0, 0};
  i32x8 z8 = {0, 0, 0, 0, 0, 0, 0, 0};
  __builtin_amdgcn_tensor_load_to_lds(g0, g1, z4, z4, z8, 0);
}
__device__ inline void wait_tensor0() {
  __builtin_amdgcn_s_wait_tensorcnt(0);
}

// ---------------- reduction helpers (wave32, 16-lane halves) ----------------
__device__ inline float halfmax16(float v) {
  #pragma unroll
  for (int m = 1; m < 16; m <<= 1) v = fmaxf(v, __shfl_xor(v, m, 32));
  return v;
}
__device__ inline float halfsum16(float v) {
  #pragma unroll
  for (int m = 1; m < 16; m <<= 1) v += __shfl_xor(v, m, 32);
  return v;
}

// ---------------- f32 -> bf16 convert (vectorized x4) ----------------
__global__ void cvt_bf16_kernel(const float* __restrict__ in,
                                __bf16* __restrict__ out, int n4) {
  int i = blockIdx.x * blockDim.x + threadIdx.x;
  if (i >= n4) return;
  float4 v = reinterpret_cast<const float4*>(in)[i];
  union { __bf16 b[4]; uint2 u; } p;
  p.b[0] = (__bf16)v.x; p.b[1] = (__bf16)v.y;
  p.b[2] = (__bf16)v.z; p.b[3] = (__bf16)v.w;
  reinterpret_cast<uint2*>(out)[i] = p.u;
}

// ---------------- WMMA bf16 GEMM:  C[M,N] = A[M,K] @ W[N,K]^T ----------------
// Block: 256 threads (8 wave32). Tile 64(M) x 128(N), K-step 32.
// Double-buffered LDS staged with async-to-LDS DMA (overlaps with WMMA).
#define GT_M 64
#define GT_N 128
#define GT_K 32

__global__ __launch_bounds__(256, 2)
void gemm_bf16_nt_kernel(const __bf16* __restrict__ A,
                         const __bf16* __restrict__ W,
                         float* __restrict__ C,
                         int M, int N, int K) {
  __shared__ __align__(16) __bf16 Al[2][GT_M][GT_K + 8];  // 80B row stride
  __shared__ __align__(16) __bf16 Bl[2][GT_N][GT_K + 8];

  const int tid  = threadIdx.x;
  const int lane = tid & 31;
  const int wave = tid >> 5;
  const int wm   = wave & 3;
  const int wn   = wave >> 2;
  const int bm   = blockIdx.x * GT_M;
  const int bn   = blockIdx.y * GT_N;
  const int hh   = lane >> 4;
  const int l16  = lane & 15;

  // Per-thread staging chunk: row r (0..63), 16B column offset cc.
  const int sr = tid >> 2;
  const int sc_ = (tid & 3) * 8;

  auto stage = [&](int bufi, int k0) {
    // A tile: 64x32 = 256 x 16B chunks, one per thread (unconditional)
    async_load_b128((unsigned)(uintptr_t)&Al[bufi][sr][sc_],
                    A + (size_t)(bm + sr) * K + k0 + sc_);
    // B tile: 128x32 = 512 chunks, two per thread (rows sr and sr+64)
    async_load_b128((unsigned)(uintptr_t)&Bl[bufi][sr][sc_],
                    W + (size_t)(bn + sr) * K + k0 + sc_);
    async_load_b128((unsigned)(uintptr_t)&Bl[bufi][sr + 64][sc_],
                    W + (size_t)(bn + sr + 64) * K + k0 + sc_);
  };

  v8f acc[4] = {};
  stage(0, 0);
  int buf = 0;

  for (int k0 = 0; k0 < K; k0 += GT_K) {
    wait_async0();          // my DMA for tile (k0) done
    __syncthreads();        // everyone's DMA done; prev-buffer reads finished
    if (k0 + GT_K < K) stage(buf ^ 1, k0 + GT_K);   // prefetch next tile

    v16bf af;               // A fragment (16x32 bf16, ISA layout)
    #pragma unroll
    for (int i = 0; i < 16; ++i) {
      int kk = (i & 7) + ((i >> 3) << 4) + (hh << 3);
      af[i] = Al[buf][wm * 16 + l16][kk];
    }
    #pragma unroll
    for (int t = 0; t < 4; ++t) {
      v16bf bfm;            // B fragment: lane = N col, (i&7) + 16*(i>>3) + 8*half = K idx
      #pragma unroll
      for (int i = 0; i < 16; ++i) {
        int kk = (i & 7) + ((i >> 3) << 4) + (hh << 3);
        bfm[i] = Bl[buf][wn * 64 + t * 16 + l16][kk];
      }
      acc[t] = __builtin_amdgcn_wmma_f32_16x16x32_bf16(
          false, af, false, bfm, (short)0, acc[t], false, false);
    }
    buf ^= 1;
  }

  #pragma unroll
  for (int t = 0; t < 4; ++t) {
    int col = bn + wn * 64 + t * 16 + l16;
    #pragma unroll
    for (int r = 0; r < 8; ++r) {
      int row = bm + wm * 16 + r + (hh << 3);
      C[(size_t)row * N + col] = acc[t][r];
    }
  }
}

// ---------------- RMSNorm + RoPE + scale, transpose to (B,NH,S,HD) bf16 -----
__global__ __launch_bounds__(256)
void norm_rope_kernel(const float* __restrict__ X,      // (B*S, NH*HD) f32
                      const float* __restrict__ cosT,   // (S, HD)
                      const float* __restrict__ sinT,   // (S, HD)
                      const float* __restrict__ scale,  // (HD)
                      __bf16* __restrict__ Out,         // (B, NH, S, HD)
                      int NH, float outScale) {
  int row  = blockIdx.x * 8 + (threadIdx.x >> 5);
  int lane = threadIdx.x & 31;
  int nh = row % NH;
  int bs = row / NH;
  int s  = bs % S_;
  int b  = bs / S_;

  const float* src = X + (size_t)bs * (NH * HD_) + nh * HD_;
  float x1 = src[lane];
  float x2 = src[lane + 32];
  float ss = x1 * x1 + x2 * x2;
  #pragma unroll
  for (int m = 1; m < 32; m <<= 1) ss += __shfl_xor(ss, m, 32);
  float r = rsqrtf(ss * (1.0f / 64.0f) + 1e-6f);
  float y1 = x1 * r * (1.0f + scale[lane]);
  float y2 = x2 * r * (1.0f + scale[lane + 32]);
  float c  = cosT[s * HD_ + lane];
  float sn = sinT[s * HD_ + lane];   // cos/sin halves are duplicated
  float o1 = (y1 * c - y2 * sn) * outScale;
  float o2 = (y2 * c + y1 * sn) * outScale;
  __bf16* dst = Out + (((size_t)(b * NH + nh)) * S_ + s) * HD_;
  dst[lane]      = (__bf16)o1;
  dst[lane + 32] = (__bf16)o2;
}

// ---------------- V: (B*S, G*HD) f32 -> (B,G,S,HD) bf16 ----------------
__global__ void v_pack_kernel(const float* __restrict__ X,
                              __bf16* __restrict__ Out) {
  size_t i = (size_t)blockIdx.x * blockDim.x + threadIdx.x;
  size_t n = (size_t)B_ * S_ * G_ * HD_;
  if (i >= n) return;
  int d = (int)(i % HD_); size_t t = i / HD_;
  int g = (int)(t % G_);  t /= G_;
  int s = (int)(t % S_);
  int b = (int)(t / S_);
  Out[(((size_t)(b * G_ + g)) * S_ + s) * HD_ + d] = (__bf16)X[i];
}

// ---------------- Flash attention (online softmax, WMMA bf16, TDM K/V) ------
// Block: 128 threads = 4 wave32; 64 query rows per block (16 per wave).
// Wave 0 drives the Tensor Data Mover: double-buffered 32-key K/V tiles.
__global__ __launch_bounds__(128)
void attn_fwd_kernel(const __bf16* __restrict__ Q,   // (B,H,S,HD)
                     const __bf16* __restrict__ Kg,  // (B,G,S,HD)
                     const __bf16* __restrict__ Vg,  // (B,G,S,HD)
                     __bf16* __restrict__ Ctx) {     // (B,S,H*HD)
  __shared__ __align__(16) __bf16 Kl[2][32][HD_ + 8];   // 144B row stride
  __shared__ __align__(16) __bf16 Vl[2][32][HD_ + 8];
  __shared__ __align__(16) __bf16 Pl[4][16][32 + 8];

  const int tid  = threadIdx.x;
  const int lane = tid & 31;
  const int wave = tid >> 5;
  const int hh   = lane >> 4;
  const int l16  = lane & 15;

  const int bh = blockIdx.y;           // b*H + h
  const int b  = bh / H_;
  const int h  = bh % H_;
  const int g  = h / (H_ / G_);
  const int q0 = blockIdx.x * 64;

  // Q fragments: rows q0+wave*16 .. +15, 2 frags over HD
  const __bf16* Qbase = Q + (((size_t)bh) * S_ + q0 + wave * 16) * HD_;
  v16bf qf[2];
  #pragma unroll
  for (int f = 0; f < 2; ++f)
    #pragma unroll
    for (int i = 0; i < 16; ++i) {
      int kk = f * 32 + (i & 7) + ((i >> 3) << 4) + (hh << 3);
      qf[f][i] = Qbase[(size_t)l16 * HD_ + kk];
    }

  const __bf16* Kbase = Kg + (((size_t)(b * G_ + g)) * S_) * HD_;
  const __bf16* Vbase = Vg + (((size_t)(b * G_ + g)) * S_) * HD_;

  v8f acc[4] = {};
  float mrow[8], lrow[8];
  #pragma unroll
  for (int r = 0; r < 8; ++r) { mrow[r] = -3.0e38f; lrow[r] = 0.0f; }

  const int nk = q0 + 64;              // causal: keys 0 .. q0+63
  if (wave == 0) {                     // prologue: TDM tile 0 into buffer 0
    tdm_load_tile32((unsigned)(uintptr_t)&Kl[0][0][0], Kbase);
    tdm_load_tile32((unsigned)(uintptr_t)&Vl[0][0][0], Vbase);
  }
  int buf = 0;

  for (int kt = 0; kt < nk; kt += 32) {
    if (wave == 0) wait_tensor0();     // tile kt landed in LDS
    __syncthreads();                   // visible to all; prev buffer free
    if (wave == 0 && kt + 32 < nk) {   // TDM prefetch next tile
      tdm_load_tile32((unsigned)(uintptr_t)&Kl[buf ^ 1][0][0],
                      Kbase + (size_t)(kt + 32) * HD_);
      tdm_load_tile32((unsigned)(uintptr_t)&Vl[buf ^ 1][0][0],
                      Vbase + (size_t)(kt + 32) * HD_);
    }

    // scores = Q @ K^T : two 16x16 f32 fragments
    v8f sc[2] = {};
    #pragma unroll
    for (int nf = 0; nf < 2; ++nf) {
      #pragma unroll
      for (int f = 0; f < 2; ++f) {
        v16bf bfm;   // B[k=hd][n=key] = K[key][hd]
        #pragma unroll
        for (int i = 0; i < 16; ++i) {
          int kk = f * 32 + (i & 7) + ((i >> 3) << 4) + (hh << 3);
          bfm[i] = Kl[buf][nf * 16 + l16][kk];
        }
        sc[nf] = __builtin_amdgcn_wmma_f32_16x16x32_bf16(
            false, qf[f], false, bfm, (short)0, sc[nf], false, false);
      }
    }

    // online softmax per row (row = r + 8*half; col = lane&15 within frag)
    #pragma unroll
    for (int r = 0; r < 8; ++r) {
      int qrow = q0 + wave * 16 + r + (hh << 3);
      float s0 = (kt + l16      <= qrow) ? sc[0][r] : -3.0e38f;
      float s1 = (kt + 16 + l16 <= qrow) ? sc[1][r] : -3.0e38f;
      float rm = halfmax16(fmaxf(s0, s1));
      float mn = fmaxf(mrow[r], rm);
      float alpha = __expf(mrow[r] - mn);
      float p0 = __expf(s0 - mn);
      float p1 = __expf(s1 - mn);
      float rs = halfsum16(p0 + p1);
      lrow[r] = lrow[r] * alpha + rs;
      mrow[r] = mn;
      #pragma unroll
      for (int t = 0; t < 4; ++t) acc[t][r] *= alpha;
      Pl[wave][r + (hh << 3)][l16]      = (__bf16)p0;
      Pl[wave][r + (hh << 3)][16 + l16] = (__bf16)p1;
    }

    // P (16x32) @ V (32x64) -> acc
    v16bf pf;
    #pragma unroll
    for (int i = 0; i < 16; ++i) {
      int kk = (i & 7) + ((i >> 3) << 4) + (hh << 3);
      pf[i] = Pl[wave][l16][kk];
    }
    #pragma unroll
    for (int t = 0; t < 4; ++t) {
      v16bf bfm;   // B[k=key][n=hd] = V[key][hd]
      #pragma unroll
      for (int i = 0; i < 16; ++i) {
        bfm[i] = Vl[buf][(i & 7) + ((i >> 3) << 4) + (hh << 3)][t * 16 + l16];
      }
      acc[t] = __builtin_amdgcn_wmma_f32_16x16x32_bf16(
          false, pf, false, bfm, (short)0, acc[t], false, false);
    }
    buf ^= 1;
  }

  // write context: (B,S,H*HD) bf16
  #pragma unroll
  for (int r = 0; r < 8; ++r) {
    int s = q0 + wave * 16 + r + (hh << 3);
    float inv = 1.0f / lrow[r];
    __bf16* dst = Ctx + ((size_t)b * S_ + s) * (size_t)(H_ * HD_) + h * HD_;
    #pragma unroll
    for (int t = 0; t < 4; ++t) dst[t * 16 + l16] = (__bf16)(acc[t][r] * inv);
  }
}

// ---------------- launcher ----------------
extern "C" void kernel_launch(void* const* d_in, const int* in_sizes, int n_in,
                              void* d_out, int out_size, void* d_ws, size_t ws_size,
                              hipStream_t stream) {
  const float* x    = (const float*)d_in[0];
  // d_in[1] = mask (bool) — causal mask computed analytically, unused
  const float* cosT = (const float*)d_in[2];
  const float* sinT = (const float*)d_in[3];
  const float* Wq   = (const float*)d_in[4];
  const float* Wk   = (const float*)d_in[5];
  const float* Wv   = (const float*)d_in[6];
  const float* Wo   = (const float*)d_in[7];
  const float* qsc  = (const float*)d_in[8];
  const float* ksc  = (const float*)d_in[9];
  float* out = (float*)d_out;

  const size_t nBS  = (size_t)B_ * S_;        // 4096
  const size_t nX   = nBS * D_;               // 8.4M
  const size_t nWq  = (size_t)H_ * HD_ * D_;  // 4.2M
  const size_t nWkv = (size_t)G_ * HD_ * D_;  // 1.05M
  const size_t nQ   = nBS * H_ * HD_;         // 8.4M
  const size_t nKV  = nBS * G_ * HD_;         // 2.1M

  char* ws = (char*)d_ws;
  size_t off = 0;
  auto alloc = [&](size_t bytes) { char* p = ws + off; off = (off + bytes + 255) & ~(size_t)255; return p; };

  __bf16* xb   = (__bf16*)alloc(nX * 2);
  __bf16* Wqb  = (__bf16*)alloc(nWq * 2);
  __bf16* Wkb  = (__bf16*)alloc(nWkv * 2);
  __bf16* Wvb  = (__bf16*)alloc(nWkv * 2);
  __bf16* Wob  = (__bf16*)alloc(nWq * 2);
  float*  Qf   = (float*)alloc(nQ * 4);
  float*  Kf   = (float*)alloc(nKV * 4);
  float*  Vf   = (float*)alloc(nKV * 4);
  __bf16* Qb   = (__bf16*)alloc(nQ * 2);
  __bf16* Kb   = (__bf16*)alloc(nKV * 2);
  __bf16* Vb   = (__bf16*)alloc(nKV * 2);
  __bf16* Ctxb = (__bf16*)alloc(nQ * 2);

  auto cvt = [&](const float* src, __bf16* dst, size_t n) {
    int n4 = (int)(n / 4);
    cvt_bf16_kernel<<<(n4 + 255) / 256, 256, 0, stream>>>(src, dst, n4);
  };
  cvt(x,  xb,  nX);
  cvt(Wq, Wqb, nWq);
  cvt(Wk, Wkb, nWkv);
  cvt(Wv, Wvb, nWkv);
  cvt(Wo, Wob, nWq);

  const int M = (int)nBS, K = D_;
  dim3 blk(256);
  gemm_bf16_nt_kernel<<<dim3(M / GT_M, (H_ * HD_) / GT_N), blk, 0, stream>>>(xb, Wqb, Qf, M, H_ * HD_, K);
  gemm_bf16_nt_kernel<<<dim3(M / GT_M, (G_ * HD_) / GT_N), blk, 0, stream>>>(xb, Wkb, Kf, M, G_ * HD_, K);
  gemm_bf16_nt_kernel<<<dim3(M / GT_M, (G_ * HD_) / GT_N), blk, 0, stream>>>(xb, Wvb, Vf, M, G_ * HD_, K);

  norm_rope_kernel<<<(int)(nBS * H_ / 8), 256, 0, stream>>>(Qf, cosT, sinT, qsc, Qb, H_, 0.125f);
  norm_rope_kernel<<<(int)(nBS * G_ / 8), 256, 0, stream>>>(Kf, cosT, sinT, ksc, Kb, G_, 1.0f);
  v_pack_kernel<<<(int)((nKV + 255) / 256), 256, 0, stream>>>(Vf, Vb);

  attn_fwd_kernel<<<dim3(S_ / 64, B_ * H_), dim3(128), 0, stream>>>(Qb, Kb, Vb, Ctxb);

  gemm_bf16_nt_kernel<<<dim3(M / GT_M, D_ / GT_N), blk, 0, stream>>>(Ctxb, Wob, out, M, D_, H_ * HD_);
}
